// MoG_part_lvl_MLP_52132313039087
// MI455X (gfx1250) — matrix operand order repaired and checked by reading, not compiled, not run on hardware
//
#include <hip/hip_runtime.h>
#include <hip/hip_bf16.h>
#include <stdint.h>

// Problem constants (match reference)
#define BB 512
#define SS 256
#define EE 32
#define NK 3

typedef __attribute__((ext_vector_type(16))) _Float16 v16h;
typedef __attribute__((ext_vector_type(8)))  float    v8f;

// ---------------------------------------------------------------------------
// Kernel 0: fold the attention weights.
//   qv = Wq@query + bq
//   u  = Wk^T qv                (score direction)     c0 = qv . bk
//   M2 = Wo @ Wv                (fused out projection) c2 = Wo@bv + bo
// Also stores M2^T as WMMA f16 B operands in contiguous k order (32x16 per N-tile):
//   lanes 0-15 : N = lane,   K = 0..15  (VGPR r holds K=2r,2r+1)
//   lanes16-31 : N = lane-16,K = 16..31
// ---------------------------------------------------------------------------
__global__ void pre_kernel(const float* __restrict__ queries,
                           const float* __restrict__ Wq, const float* __restrict__ bq,
                           const float* __restrict__ Wk, const float* __restrict__ bk,
                           const float* __restrict__ Wv, const float* __restrict__ bv,
                           const float* __restrict__ Wo, const float* __restrict__ bo,
                           float* __restrict__ u_ws, float* __restrict__ c0_ws,
                           float* __restrict__ c2_ws, _Float16* __restrict__ b_ws) {
  const int ki = blockIdx.x;          // 0..2
  const int lane = threadIdx.x;       // 32 threads (one wave)
  const int E2 = EE * EE;
  __shared__ float qv[EE];
  __shared__ float M2[EE][EE];

  // qv[f] = bq[f] + sum_e queries[e] * Wq[f,e]
  {
    float acc = bq[ki * EE + lane];
    for (int e = 0; e < EE; ++e)
      acc += queries[ki * EE + e] * Wq[ki * E2 + lane * EE + e];
    qv[lane] = acc;
  }
  __syncthreads();

  // u[e] = sum_f qv[f] * Wk[f,e]
  {
    float acc = 0.f;
    for (int f = 0; f < EE; ++f)
      acc += qv[f] * Wk[ki * E2 + f * EE + lane];
    u_ws[ki * EE + lane] = acc;
  }
  if (lane == 0) {
    float c0 = 0.f;
    for (int f = 0; f < EE; ++f) c0 += qv[f] * bk[ki * EE + f];
    c0_ws[ki] = c0;
  }
  // M2[g][e] = sum_f Wo[g,f]*Wv[f,e]
  for (int g = 0; g < EE; ++g) {
    float acc = 0.f;
    for (int f = 0; f < EE; ++f)
      acc += Wo[ki * E2 + g * EE + f] * Wv[ki * E2 + f * EE + lane];
    M2[g][lane] = acc;
  }
  // c2[g] = Wo@bv + bo
  {
    float acc = bo[ki * EE + lane];
    for (int f = 0; f < EE; ++f)
      acc += Wo[ki * E2 + lane * EE + f] * bv[ki * EE + f];
    c2_ws[ki * EE + lane] = acc;
  }
  __syncthreads();

  // Emit B operand tiles (N-tile 0: g=0..15, N-tile 1: g=16..31), K = channel e.
  for (int tile = 0; tile < 2; ++tile) {
    const int g  = tile * 16 + (lane & 15);
    const int kb = (lane & 16) ? 16 : 0;
    _Float16* dst = b_ws + ((size_t)(ki * 2 + tile) * 32 + lane) * 16;
    for (int j = 0; j < 16; ++j) dst[j] = (_Float16)M2[g][kb + j];
  }
}

// ---------------------------------------------------------------------------
// Kernel 1: per-particle kinematics (eta, phi, energy)
// v layout: (B, E, S) -> v[b*E*S + e*S + s]
// ---------------------------------------------------------------------------
__global__ void geom_kernel(const float* __restrict__ v,
                            float* __restrict__ eta_ws,
                            float* __restrict__ phi_ws,
                            float* __restrict__ en_ws) {
  const int g = blockIdx.x * blockDim.x + threadIdx.x;   // b*S + s
  const int b = g >> 8, s = g & 255;
  const float* vb = v + (size_t)b * EE * SS;
  const float px = vb[0 * SS + s];
  const float py = vb[1 * SS + s];
  const float pz = vb[2 * SS + s];
  const float en = vb[3 * SS + s];
  const float pT = sqrtf(px * px + py * py);
  eta_ws[g] = 0.5f * logf((pT + pz) / (pT - pz));
  phi_ws[g] = atan2f(py, px);
  en_ws[g]  = en;
}

// ---------------------------------------------------------------------------
// Kernel 2: KNN top-16 (sorted ascending by distance, stable lowest-index tie
// break via packed (floatbits<<32)|idx u64 keys -> matches lax.top_k).
// One block per (jet, row): blockIdx.x = b*S + i, 256 threads = columns.
// ---------------------------------------------------------------------------
__global__ void knn_kernel(const float* __restrict__ eta_ws,
                           const float* __restrict__ phi_ws,
                           const float* __restrict__ en_ws,
                           int* __restrict__ knn_ws) {
  const int b = blockIdx.x >> 8, i = blockIdx.x & 255;
  const int j = threadIdx.x;
  __shared__ unsigned long long keys[SS];
  __shared__ int winner;

  const float PI_F  = 3.14159265358979323846f;
  const float TWOPI = 6.28318530717958647692f;
  const int base = b * SS;

  const float ei  = eta_ws[base + i];
  const float pi_ = phi_ws[base + i];
  const float eni = en_ws[base + i];

  const float deta = ei - eta_ws[base + j];
  float dphi = pi_ - phi_ws[base + j] + PI_F;
  dphi = fmodf(dphi, TWOPI);
  if (dphi < 0.f) dphi += TWOPI;
  dphi -= PI_F;
  const float ed = fabsf(eni - en_ws[base + j]);
  const float d  = ed * sqrtf(deta * deta + dphi * dphi);   // >= 0, finite

  const unsigned long long mykey =
      (((unsigned long long)__float_as_uint(d)) << 32) | (unsigned)j;
  bool alive = true;

  for (int t = 0; t < 16; ++t) {
    keys[j] = alive ? mykey : 0xFFFFFFFFFFFFFFFFull;
    __syncthreads();
    for (int s2 = 128; s2 > 0; s2 >>= 1) {
      if (j < s2) {
        unsigned long long o = keys[j + s2];
        if (o < keys[j]) keys[j] = o;
      }
      __syncthreads();
    }
    if (j == 0) {
      winner = (int)(keys[0] & 0xFFFFFFFFu);
      knn_ws[(size_t)blockIdx.x * 16 + t] = winner;
    }
    __syncthreads();
    if (j == winner) alive = false;
  }
}

// ---------------------------------------------------------------------------
// Kernel 3: attention pooling + fused WMMA out-projection.
// One block (256 thr = 8 waves) per jet b. Lane owns one particle: computes
// masked scores -> softmax -> pooled p (all in VGPRs, fully unrolled), stages
// p as f16 in LDS, reloads in the 16-bit WMMA A layout, then
// v_wmma_f32_16x16x32_f16 applies M2 (2 M-tiles x 2 N-tiles x 3 heads).
// ---------------------------------------------------------------------------
__global__ void __launch_bounds__(256)
attn_kernel(const float* __restrict__ v, const uint8_t* __restrict__ maskp,
            const int* __restrict__ knn_ws, const float* __restrict__ u_ws,
            const float* __restrict__ c0_ws, const float* __restrict__ c2_ws,
            const _Float16* __restrict__ b_ws, float* __restrict__ out) {
  __shared__ float    sv[SS * 33];        // jet features, row-padded (33 floats)
  __shared__ int      pads[SS];           // 1 = padded particle
  __shared__ float    uu[NK * EE];
  __shared__ float    c2s[NK * EE];
  __shared__ float    c0s[NK];
  __shared__ _Float16 pbuf[8 * 32 * 34];  // per-wave pooled vectors, pad 34

  const int b = blockIdx.x;
  const int tid = threadIdx.x;

  const float* vb = v + (size_t)b * EE * SS;
  #pragma unroll
  for (int e = 0; e < EE; ++e)
    sv[tid * 33 + e] = vb[e * SS + tid];          // coalesced over tid
  pads[tid] = (maskp[b * SS + tid] == 0) ? 1 : 0;
  if (tid < NK * EE) { uu[tid] = u_ws[tid]; c2s[tid] = c2_ws[tid]; }
  if (tid < NK)      { c0s[tid] = c0_ws[tid]; }
  __syncthreads();

  const int w    = tid >> 5;
  const int lane = tid & 31;
  const int sOwn = w * 32 + lane;                 // particle owned in phase 1
  const int hi   = (lane >> 4) & 1;

  int nidx[16];
  const int* kr = knn_ws + ((size_t)b * SS + sOwn) * 16;
  #pragma unroll
  for (int n = 0; n < 16; ++n) nidx[n] = kr[n];

  const float scale = 0.17677669529663687f;       // 1/sqrt(32)
  _Float16* pb = pbuf + w * (32 * 34);

  #pragma unroll
  for (int ki = 0; ki < NK; ++ki) {
    const int k = (ki == 0) ? 4 : ((ki == 1) ? 8 : 16);

    // ---- masked scores
    float sc[16];
    #pragma unroll
    for (int n = 0; n < 16; ++n) {
      if (n >= k) break;
      const int r = nidx[n] * 33;
      float acc = 0.f;
      #pragma unroll
      for (int e = 0; e < EE; ++e) acc += uu[ki * EE + e] * sv[r + e];
      acc = (acc + c0s[ki]) * scale;
      sc[n] = pads[nidx[n]] ? -1e9f : acc;
    }
    // ---- softmax (per lane, serial over k)
    float m = sc[0];
    #pragma unroll
    for (int n = 1; n < 16; ++n) { if (n >= k) break; m = fmaxf(m, sc[n]); }
    float den = 0.f;
    #pragma unroll
    for (int n = 0; n < 16; ++n) { if (n >= k) break; sc[n] = __expf(sc[n] - m); den += sc[n]; }
    const float inv = 1.f / den;
    // ---- pooled vector p
    float p[EE];
    #pragma unroll
    for (int e = 0; e < EE; ++e) p[e] = 0.f;
    #pragma unroll
    for (int n = 0; n < 16; ++n) {
      if (n >= k) break;
      const float wgt = sc[n] * inv;
      const int r = nidx[n] * 33;
      #pragma unroll
      for (int e = 0; e < EE; ++e) p[e] += wgt * sv[r + e];
    }
    // ---- stage p (f16) for the WMMA A operand
    #pragma unroll
    for (int e = 0; e < EE; ++e) pb[lane * 34 + e] = (_Float16)p[e];
    asm volatile("s_wait_dscnt 0" ::: "memory");

    // ---- B operand tiles (precomputed layout, uniform across blocks)
    v16h b0, b1;
    const _Float16* bp0 = b_ws + ((size_t)(ki * 2 + 0) * 32 + lane) * 16;
    const _Float16* bp1 = b_ws + ((size_t)(ki * 2 + 1) * 32 + lane) * 16;
    #pragma unroll
    for (int jj = 0; jj < 16; ++jj) { b0[jj] = bp0[jj]; b1[jj] = bp1[jj]; }

    #pragma unroll
    for (int mt = 0; mt < 2; ++mt) {
      // A operand, contiguous k order (16x32): lanes 0-15 row M, K{0..15};
      // lanes 16-31 same rows, K{16..31}, packed pairs per VGPR.
      const int rowbase = (mt * 16 + (lane & 15)) * 34;
      v16h a;
      #pragma unroll
      for (int r = 0; r < 8; ++r) {
        const int Kp = 2 * r + (hi ? 16 : 0);
        a[2 * r]     = pb[rowbase + Kp];
        a[2 * r + 1] = pb[rowbase + Kp + 1];
      }
      #pragma unroll
      for (int nt = 0; nt < 2; ++nt) {
        v8f c = {0.f, 0.f, 0.f, 0.f, 0.f, 0.f, 0.f, 0.f};
        c = __builtin_amdgcn_wmma_f32_16x16x32_f16(
                false, a, false, (nt ? b1 : b0), (short)0, c, false, false);
        const int g = nt * 16 + (lane & 15);
        const float bias = c2s[ki * EE + g];
        #pragma unroll
        for (int r = 0; r < 8; ++r) {
          const int sOut = w * 32 + mt * 16 + r + 8 * hi;
          float val = c[r] + bias;
          if (pads[sOut]) val = 0.f;
          // out layout: (S, B, NK*E)
          out[(size_t)sOut * (BB * NK * EE) + (size_t)b * (NK * EE) + ki * EE + g] = val;
        }
      }
    }
  }
}

// ---------------------------------------------------------------------------
// Launcher. Inputs: 0=x(unused) 1=v 2=mask(u8) 3=queries 4=Wq 5=bq 6=Wk 7=bk
//                   8=Wv 9=bv 10=Wo 11=bo
// Workspace (~10 MB): eta|phi|en (B*S f32 each), knn idx (B*S*16 i32),
//                     folded weights.
// ---------------------------------------------------------------------------
extern "C" void kernel_launch(void* const* d_in, const int* in_sizes, int n_in,
                              void* d_out, int out_size, void* d_ws, size_t ws_size,
                              hipStream_t stream) {
  (void)in_sizes; (void)n_in; (void)out_size; (void)ws_size;
  const float*   v       = (const float*)d_in[1];
  const uint8_t* mask    = (const uint8_t*)d_in[2];
  const float*   queries = (const float*)d_in[3];
  const float*   Wq = (const float*)d_in[4];
  const float*   bq = (const float*)d_in[5];
  const float*   Wk = (const float*)d_in[6];
  const float*   bk = (const float*)d_in[7];
  const float*   Wv = (const float*)d_in[8];
  const float*   bv = (const float*)d_in[9];
  const float*   Wo = (const float*)d_in[10];
  const float*   bo = (const float*)d_in[11];

  const int NP = BB * SS;                       // 131072 particles
  float* eta_ws = (float*)d_ws;
  float* phi_ws = eta_ws + NP;
  float* en_ws  = phi_ws + NP;
  int*   knn_ws = (int*)(en_ws + NP);           // NP * 16 ints
  float* u_ws   = (float*)(knn_ws + (size_t)NP * 16);
  float* c0_ws  = u_ws + NK * EE;
  float* c2_ws  = c0_ws + 4;
  _Float16* b_ws = (_Float16*)(c2_ws + NK * EE);

  pre_kernel<<<NK, 32, 0, stream>>>(queries, Wq, bq, Wk, bk, Wv, bv, Wo, bo,
                                    u_ws, c0_ws, c2_ws, b_ws);
  geom_kernel<<<NP / 256, 256, 0, stream>>>(v, eta_ws, phi_ws, en_ws);
  knn_kernel<<<NP, 256, 0, stream>>>(eta_ws, phi_ws, en_ws, knn_ws);
  attn_kernel<<<BB, 256, 0, stream>>>(v, mask, knn_ws, u_ws, c0_ws, c2_ws,
                                      b_ws, (float*)d_out);
}
